// MultiQueryAttention_12592844112129
// MI455X (gfx1250) — matrix operand order repaired and checked by reading, not compiled, not run on hardware
//
#include <hip/hip_runtime.h>

#define DIM   1024
#define HEADS 16
#define HD    64
#define BATCH 2
#define SEQ   2048
#define ROWS  (BATCH * SEQ)       // 4096
#define SCALE 0.125f              // 64^-0.5

typedef __bf16 bf16;
typedef __attribute__((ext_vector_type(16))) __bf16 v16bf;
typedef __attribute__((ext_vector_type(8)))  __bf16 v8bf;
typedef __attribute__((ext_vector_type(8)))  float  v8f;
typedef __attribute__((ext_vector_type(4)))  float  v4f;
typedef __attribute__((ext_vector_type(4)))  unsigned int u32x4;
typedef __attribute__((ext_vector_type(8)))  int  i32x8;
typedef __attribute__((ext_vector_type(4)))  int  i32x4;

union Frag { v16bf v; v8bf h[2]; };

__device__ __forceinline__ v8f wmma_bf16(v16bf a, v16bf b, v8f c) {
  return __builtin_amdgcn_wmma_f32_16x16x32_bf16(
      false, a, false, b, (short)0, c, false, false);
}

// CDNA5 async copy: global -> LDS, 16B per lane, no VGPR data round-trip.
// GV addressing: 64-bit address in a VGPR pair, saddr = off. Tracked by ASYNCcnt.
__device__ __forceinline__ void async_copy_b128(void* lds_dst, const void* gsrc) {
  unsigned int l = (unsigned int)(unsigned long long)lds_dst;  // LDS offset = addr[31:0]
  asm volatile("global_load_async_to_lds_b128 %0, %1, off"
               :: "v"(l), "v"(gsrc) : "memory");
}
__device__ __forceinline__ void wait_async0() {
  asm volatile("s_wait_asynccnt 0x0" ::: "memory");
}

#if defined(__has_builtin)
#if __has_builtin(__builtin_amdgcn_tensor_load_to_lds) && \
    __has_builtin(__builtin_amdgcn_s_wait_tensorcnt)
#define HAVE_TDM 1
#endif
#endif

#ifdef HAVE_TDM
// TDM: DMA a contiguous region (n8 x 8-byte elements) from global into LDS.
// D# per CDNA5 ISA 8.3/8.4: group0 = {count, lds_addr, global_addr, type=2};
// group1 = {data_size=8B, tensor_dim0=tile_dim0=n8, dim1=1, stride0=n8}.
// This toolchain's builtin is the 6-arg form (adds an i32x8 group before cpol).
__device__ __forceinline__ void tdm_load_contig(unsigned int lds_addr,
                                                const void* gsrc,
                                                unsigned int n8) {
  unsigned long long ga = (unsigned long long)gsrc;
  u32x4 g0;
  g0[0] = 1u;                                             // count=1 (valid D#)
  g0[1] = lds_addr;                                       // bits 63:32
  g0[2] = (unsigned int)ga;                               // global_addr[31:0]
  g0[3] = (unsigned int)((ga >> 32) & 0x01FFFFFFu)        // global_addr[56:32]
        | (2u << 30);                                     // type = 2 ("image")
  i32x8 g1;
  g1[0] = (int)(3u << 16);                                // data_size=3 (8B), mask=0
  g1[1] = (int)((n8 & 0xFFFFu) << 16);                    // tensor_dim0[15:0]
  g1[2] = (int)(((n8 >> 16) & 0xFFFFu) | (1u << 16));     // dim0[31:16] | tensor_dim1=1
  g1[3] = (int)((n8 & 0xFFFFu) << 16);                    // tile_dim0 = n8
  g1[4] = 1;                                              // tile_dim1 = 1
  g1[5] = (int)n8;                                        // tensor_dim0_stride[31:0]
  g1[6] = 0;
  g1[7] = 0;
  i32x4 z4 = {};
  i32x8 z8 = {};
  __builtin_amdgcn_tensor_load_to_lds(g0, g1, z4, z4, z8, 0);
}
#endif

// ---------------------------------------------------------------------------
// C_bf16[M x Nw] = A_f32[M x 1024] * W_f32[1024 x Nw]
// block: 256 thr (8 waves), tile 128(M) x 64(N), K-step 32.
// ---------------------------------------------------------------------------
__global__ __launch_bounds__(256) void gemm_f32_to_bf16(
    const float* __restrict__ A, const float* __restrict__ W,
    bf16* __restrict__ C, int Nw)
{
  __shared__ float As[128][40];   // stride 160B (16B/32B aligned rows)
  __shared__ float Bt[64][40];    // W tile stored transposed: Bt[n][k]

  const int tid  = threadIdx.x;
  const int lane = tid & 31, wid = tid >> 5;
  const int hi   = lane >> 4, ln = lane & 15;
  const int wm   = wid >> 1,  wn = wid & 1;
  const int m0   = blockIdx.x * 128;
  const int n0   = blockIdx.y * 64;

  v8f acc[2][2] = {};

  for (int k0 = 0; k0 < DIM; k0 += 32) {
    __syncthreads();
    // stage A: 128x32 f32 via async global->LDS (4 x b128 per thread)
#pragma unroll
    for (int i = 0; i < 4; i++) {
      int idx = tid + 256 * i;                 // 0..1023 float4 slots
      int r = idx >> 3, c = (idx & 7) << 2;
      async_copy_b128(&As[r][c], A + (size_t)(m0 + r) * DIM + k0 + c);
    }
    // stage W transposed: 32x64 -> Bt[n][k] (needs lane shuffle -> ds stores)
#pragma unroll
    for (int i = 0; i < 8; i++) {
      int idx = tid + 256 * i;
      int kk = idx >> 6, nn = idx & 63;
      Bt[nn][kk] = W[(size_t)(k0 + kk) * Nw + n0 + nn];
    }
    wait_async0();
    __syncthreads();

    Frag af[2], bfr[2];
#pragma unroll
    for (int mt = 0; mt < 2; mt++) {           // A frag: K(e)=((e>>3)<<4)+(e&7)+8*hi
      int row = wm * 32 + mt * 16 + ln;
      v8f t0 = *(const v8f*)&As[row][hi * 8];
      v8f t1 = *(const v8f*)&As[row][16 + hi * 8];
      v8bf c0, c1;
#pragma unroll
      for (int e = 0; e < 8; e++) { c0[e] = (bf16)t0[e]; c1[e] = (bf16)t1[e]; }
      af[mt].h[0] = c0; af[mt].h[1] = c1;
    }
#pragma unroll
    for (int nt = 0; nt < 2; nt++) {           // B frag: K(e)=((e>>3)<<4)+(e&7)+8*hi
      int col = wn * 32 + nt * 16 + ln;
      v8f t0 = *(const v8f*)&Bt[col][hi * 8];
      v8f t1 = *(const v8f*)&Bt[col][16 + hi * 8];
      v8bf c0, c1;
#pragma unroll
      for (int e = 0; e < 8; e++) { c0[e] = (bf16)t0[e]; c1[e] = (bf16)t1[e]; }
      bfr[nt].h[0] = c0; bfr[nt].h[1] = c1;
    }
#pragma unroll
    for (int mt = 0; mt < 2; mt++)
#pragma unroll
      for (int nt = 0; nt < 2; nt++)
        acc[mt][nt] = wmma_bf16(af[mt].v, bfr[nt].v, acc[mt][nt]);
  }

#pragma unroll
  for (int mt = 0; mt < 2; mt++)
#pragma unroll
    for (int nt = 0; nt < 2; nt++)
#pragma unroll
      for (int r = 0; r < 8; r++) {            // C: (M=r+8*hi, N=ln)
        int row = m0 + wm * 32 + mt * 16 + r + hi * 8;
        int col = n0 + wn * 32 + nt * 16 + ln;
        C[(size_t)row * Nw + col] = (bf16)acc[mt][nt][r];
      }
}

// ---------------------------------------------------------------------------
// Flash MQA: 8 waves/block, each owns a 16-row q tile (same batch),
// K chunk staged by the Tensor Data Mover, V transposed via ds stores.
// ---------------------------------------------------------------------------
__global__ __launch_bounds__(256) void mqa_flash(
    const bf16* __restrict__ qbuf, const bf16* __restrict__ kbuf,
    const bf16* __restrict__ vbuf, bf16* __restrict__ obuf)
{
  __shared__ bf16 Ks[32][64];       // K chunk row-major (rows contiguous: 4KB)
  __shared__ bf16 Vt[64][32];       // V chunk transposed: Vt[d][kv]
  __shared__ bf16 Ps[8][16][32];    // per-wave P bounce buffer

  const int tid  = threadIdx.x;
  const int lane = tid & 31, wid = tid >> 5;
  const int hi   = lane >> 4, ln = lane & 15;
  const int b    = blockIdx.x >> 8;            // 256 blocks per batch
  const int loc  = blockIdx.x & 255;
  const int idx  = loc * 8 + wid;              // 0..2047
  const int h    = idx >> 7;                   // head 0..15
  const int qrow0 = (idx & 127) * 16;          // q tile base row

  // q A-fragments for d=0..31 and d=32..63 (held in VGPRs all loop)
  Frag qA[2];
  {
    const bf16* qp = qbuf + ((size_t)(b * SEQ + qrow0 + ln)) * DIM + h * HD;
#pragma unroll
    for (int s = 0; s < 2; s++) {
      qA[s].h[0] = *(const v8bf*)(qp + 32 * s + hi * 8);
      qA[s].h[1] = *(const v8bf*)(qp + 32 * s + 16 + hi * 8);
    }
  }

  v8f   o[4] = {};
  float m[8], l[8];
#pragma unroll
  for (int r = 0; r < 8; r++) { m[r] = -1e30f; l[r] = 0.0f; }

  const bf16* kb = kbuf + (size_t)b * SEQ * HD;
  const bf16* vb = vbuf + (size_t)b * SEQ * HD;

  for (int kv0 = 0; kv0 < SEQ; kv0 += 32) {
    __syncthreads();
    // stage K: 32x64 bf16 = 4KB contiguous -> one TDM descriptor from wave 0
#ifdef HAVE_TDM
    if (wid == 0) {
      tdm_load_contig((unsigned int)(unsigned long long)&Ks[0][0],
                      kb + (size_t)kv0 * HD, 512 /* 8B elements */);
    }
#else
    {
      int r = tid >> 3, c = (tid & 7) << 3;
      async_copy_b128(&Ks[r][c], kb + (size_t)(kv0 + r) * HD + c);
    }
#endif
#pragma unroll
    for (int i = 0; i < 8; i++) {   // stage V transposed (lane shuffle -> ds)
      int id2 = tid + 256 * i;
      int kv = id2 >> 6, d = id2 & 63;
      Vt[d][kv] = vb[(size_t)(kv0 + kv) * HD + d];
    }
    if (kv0 + 32 < SEQ) {           // pull next chunk toward L2/L0
      __builtin_prefetch(kb + (size_t)(kv0 + 32) * HD, 0, 1);
      __builtin_prefetch(vb + (size_t)(kv0 + 32) * HD, 0, 1);
    }
#ifdef HAVE_TDM
    if (wid == 0) __builtin_amdgcn_s_wait_tensorcnt(0);
#else
    wait_async0();
#endif
    __syncthreads();

    // S = q . k^T : two 16-col tiles, K(head-dim)=64 -> 2 WMMA each
    v8f s0 = {}, s1 = {};
#pragma unroll
    for (int s = 0; s < 2; s++) {
      Frag k0f, k1f;
      k0f.h[0] = *(const v8bf*)&Ks[ln][32 * s + hi * 8];
      k0f.h[1] = *(const v8bf*)&Ks[ln][32 * s + 16 + hi * 8];
      k1f.h[0] = *(const v8bf*)&Ks[ln + 16][32 * s + hi * 8];
      k1f.h[1] = *(const v8bf*)&Ks[ln + 16][32 * s + 16 + hi * 8];
      s0 = wmma_bf16(qA[s].v, k0f.v, s0);
      s1 = wmma_bf16(qA[s].v, k1f.v, s1);
    }
    s0 *= SCALE; s1 *= SCALE;

    // online softmax: row = r + 8*hi lives in one 16-lane half
    float alpha[8];
#pragma unroll
    for (int r = 0; r < 8; r++) {
      float mx = fmaxf(s0[r], s1[r]);
#pragma unroll
      for (int off = 8; off; off >>= 1) mx = fmaxf(mx, __shfl_xor(mx, off, 16));
      float mn = fmaxf(m[r], mx);
      alpha[r] = __expf(m[r] - mn);
      m[r] = mn;
      float p0 = __expf(s0[r] - mn);
      float p1 = __expf(s1[r] - mn);
      s0[r] = p0; s1[r] = p1;
      float rs = p0 + p1;
#pragma unroll
      for (int off = 8; off; off >>= 1) rs += __shfl_xor(rs, off, 16);
      l[r] = l[r] * alpha[r] + rs;
    }
#pragma unroll
    for (int dt = 0; dt < 4; dt++)
#pragma unroll
      for (int r = 0; r < 8; r++) o[dt][r] *= alpha[r];

    // bounce P through per-wave LDS: C-layout -> A-layout (bf16)
#pragma unroll
    for (int r = 0; r < 8; r++) {
      Ps[wid][r + hi * 8][ln]      = (bf16)s0[r];
      Ps[wid][r + hi * 8][16 + ln] = (bf16)s1[r];
    }
    asm volatile("s_wait_dscnt 0x0" ::: "memory");
    Frag pA;
    pA.h[0] = *(const v8bf*)&Ps[wid][ln][hi * 8];
    pA.h[1] = *(const v8bf*)&Ps[wid][ln][16 + hi * 8];

    // O += P . V : 4 d-tiles, K(kv)=32 -> 1 WMMA each
#pragma unroll
    for (int dt = 0; dt < 4; dt++) {
      Frag vB;
      vB.h[0] = *(const v8bf*)&Vt[dt * 16 + ln][hi * 8];
      vB.h[1] = *(const v8bf*)&Vt[dt * 16 + ln][16 + hi * 8];
      o[dt] = wmma_bf16(pA.v, vB.v, o[dt]);
    }
  }

  // normalize and store attention output (bf16) for the final projection
#pragma unroll
  for (int dt = 0; dt < 4; dt++)
#pragma unroll
    for (int r = 0; r < 8; r++) {
      int n = qrow0 + r + hi * 8;
      float val = o[dt][r] / l[r];
      obuf[((size_t)(b * SEQ + n)) * DIM + h * HD + dt * 16 + ln] = (bf16)val;
    }
}

// ---------------------------------------------------------------------------
// out_f32[M x 1024] = A_bf16[M x 1024] * Wp_f32[1024 x 1024] + bp
// ---------------------------------------------------------------------------
__global__ __launch_bounds__(256) void gemm_bf16_bias_f32(
    const bf16* __restrict__ A, const float* __restrict__ W,
    const float* __restrict__ bias, float* __restrict__ Cout)
{
  __shared__ bf16  As[128][40];   // 80B row stride (16B aligned)
  __shared__ float Bt[64][40];

  const int tid  = threadIdx.x;
  const int lane = tid & 31, wid = tid >> 5;
  const int hi   = lane >> 4, ln = lane & 15;
  const int wm   = wid >> 1,  wn = wid & 1;
  const int m0   = blockIdx.x * 128;
  const int n0   = blockIdx.y * 64;

  v8f acc[2][2] = {};

  for (int k0 = 0; k0 < DIM; k0 += 32) {
    __syncthreads();
#pragma unroll
    for (int i = 0; i < 2; i++) {   // A tile bf16: async global->LDS, 2 x b128
      int idx = tid + 256 * i;
      int r = idx >> 2, c = (idx & 3) << 3;
      async_copy_b128(&As[r][c], A + (size_t)(m0 + r) * DIM + k0 + c);
    }
#pragma unroll
    for (int i = 0; i < 8; i++) {   // W tile transposed
      int idx = tid + 256 * i;
      int kk = idx >> 6, nn = idx & 63;
      Bt[nn][kk] = W[(size_t)(k0 + kk) * DIM + n0 + nn];
    }
    wait_async0();
    __syncthreads();

    Frag af[2], bfr[2];
#pragma unroll
    for (int mt = 0; mt < 2; mt++) {   // bf16 A: direct vector loads, no cvt
      int row = wm * 32 + mt * 16 + ln;
      af[mt].h[0] = *(const v8bf*)&As[row][hi * 8];
      af[mt].h[1] = *(const v8bf*)&As[row][16 + hi * 8];
    }
#pragma unroll
    for (int nt = 0; nt < 2; nt++) {
      int col = wn * 32 + nt * 16 + ln;
      v8f t0 = *(const v8f*)&Bt[col][hi * 8];
      v8f t1 = *(const v8f*)&Bt[col][16 + hi * 8];
      v8bf c0, c1;
#pragma unroll
      for (int e = 0; e < 8; e++) { c0[e] = (bf16)t0[e]; c1[e] = (bf16)t1[e]; }
      bfr[nt].h[0] = c0; bfr[nt].h[1] = c1;
    }
#pragma unroll
    for (int mt = 0; mt < 2; mt++)
#pragma unroll
      for (int nt = 0; nt < 2; nt++)
        acc[mt][nt] = wmma_bf16(af[mt].v, bfr[nt].v, acc[mt][nt]);
  }

#pragma unroll
  for (int mt = 0; mt < 2; mt++)
#pragma unroll
    for (int nt = 0; nt < 2; nt++)
#pragma unroll
      for (int r = 0; r < 8; r++) {
        int row = m0 + wm * 32 + mt * 16 + r + hi * 8;
        int col = n0 + wn * 32 + nt * 16 + ln;
        Cout[(size_t)row * DIM + col] = acc[mt][nt][r] + bias[col];
      }
}

// ---------------------------------------------------------------------------
extern "C" void kernel_launch(void* const* d_in, const int* in_sizes, int n_in,
                              void* d_out, int out_size, void* d_ws, size_t ws_size,
                              hipStream_t stream)
{
  const float* x  = (const float*)d_in[0];
  const float* Wq = (const float*)d_in[1];
  const float* Wk = (const float*)d_in[2];
  const float* Wv = (const float*)d_in[3];
  const float* Wp = (const float*)d_in[4];
  const float* bp = (const float*)d_in[5];
  float* out = (float*)d_out;

  // workspace layout (bf16): q[4096x1024] | k[4096x64] | v[4096x64] | attnO[4096x1024]
  bf16* qbuf = (bf16*)d_ws;
  bf16* kbuf = qbuf + (size_t)ROWS * DIM;
  bf16* vbuf = kbuf + (size_t)ROWS * HD;
  bf16* obuf = vbuf + (size_t)ROWS * HD;

  dim3 blk(256);
  gemm_f32_to_bf16<<<dim3(ROWS / 128, DIM / 64), blk, 0, stream>>>(x, Wq, qbuf, DIM);
  gemm_f32_to_bf16<<<dim3(ROWS / 128, 1),        blk, 0, stream>>>(x, Wk, kbuf, HD);
  gemm_f32_to_bf16<<<dim3(ROWS / 128, 1),        blk, 0, stream>>>(x, Wv, vbuf, HD);
  mqa_flash<<<dim3(BATCH * 256), blk, 0, stream>>>(qbuf, kbuf, vbuf, obuf);
  gemm_bf16_bias_f32<<<dim3(ROWS / 128, DIM / 64), blk, 0, stream>>>(obuf, Wp, bp, out);
}
